// cRNNv2_17386027615005
// MI455X (gfx1250) — compile-verified, matrix operand order checked by reading, not run on hardware
//
#include <hip/hip_runtime.h>
#include <hip/hip_bf16.h>

typedef __attribute__((ext_vector_type(16))) __bf16 v16bf;
typedef __attribute__((ext_vector_type(8)))  __bf16 v8bf;
typedef __attribute__((ext_vector_type(8)))  float  v8f;

union FragU { v16bf v; v8bf h[2]; };

// Load a 16x32 bf16 fragment (A layout; B 32x16 uses the mirrored identical
// per-lane pattern) from row-major K-contiguous memory.
// lane 0-15  : row = base_row+lane,    K in [k0, k0+8) and [k0+16, k0+24)
// lane 16-31 : row = base_row+lane-16, K in [k0+8,k0+16) and [k0+24, k0+32)
__device__ __forceinline__ v16bf load_frag_nt(const __bf16* __restrict__ base,
                                              int row, int ld, int k0, int lane) {
    const int r  = row + (lane & 15);
    const int kk = k0 + ((lane >> 4) << 3);
    FragU f;
    f.h[0] = *reinterpret_cast<const v8bf*>(base + (size_t)r * ld + kk);
    f.h[1] = *reinterpret_cast<const v8bf*>(base + (size_t)r * ld + kk + 16);
    return f.v;
}

// C[M,N] (f32) = A[M,K] (bf16, row-major) * B[N,K]^T (bf16, row-major)
// Block: 256 threads = 8 waves, block tile 128x128, wave tile 64x32.
__global__ __launch_bounds__(256) void gemm_nt_bf16(
    const __bf16* __restrict__ A, const __bf16* __restrict__ B,
    float* __restrict__ C, int M, int N, int K)
{
    const int lane = threadIdx.x & 31;
    const int wave = threadIdx.x >> 5;   // 0..7
    const int wm   = wave >> 2;          // 0..1  (M direction)
    const int wn   = wave & 3;           // 0..3  (N direction)
    const int m0   = blockIdx.y * 128 + wm * 64;
    const int n0   = blockIdx.x * 128 + wn * 32;

    v8f acc[4][2] = {};

    for (int k0 = 0; k0 < K; k0 += 32) {
        v16bf b0 = load_frag_nt(B, n0,      K, k0, lane);
        v16bf b1 = load_frag_nt(B, n0 + 16, K, k0, lane);
#pragma unroll
        for (int fm = 0; fm < 4; ++fm) {
            v16bf a = load_frag_nt(A, m0 + fm * 16, K, k0, lane);
            acc[fm][0] = __builtin_amdgcn_wmma_f32_16x16x32_bf16(
                false, a, false, b0, (short)0, acc[fm][0], false, false);
            acc[fm][1] = __builtin_amdgcn_wmma_f32_16x16x32_bf16(
                false, a, false, b1, (short)0, acc[fm][1], false, false);
        }
    }

    // C/D layout: VGPR i -> M = m0+fm*16 + i + (lane>>4)*8, N = n0+fn*16 + (lane&15)
    const int half = lane >> 4;
    const int ln   = lane & 15;
#pragma unroll
    for (int fm = 0; fm < 4; ++fm)
#pragma unroll
        for (int fn = 0; fn < 2; ++fn)
#pragma unroll
            for (int i = 0; i < 8; ++i) {
                int r = m0 + fm * 16 + half * 8 + i;
                int c = n0 + fn * 16 + ln;
                C[(size_t)r * N + c] = acc[fm][fn][i];
            }
}

// f32 -> packed bf16 (with row stride + column offset, to slice W_ih_dec[:, :512])
__global__ void cvt_bf16(const float* __restrict__ src, __bf16* __restrict__ dst,
                         int rows, int cols, int ld, int coff) {
    int idx = blockIdx.x * blockDim.x + threadIdx.x;
    if (idx >= rows * cols) return;
    int r = idx / cols, c = idx % cols;
    dst[idx] = (__bf16)src[(size_t)r * ld + coff + c];
}

// GRU gate fusion: one thread per (b, hh).
// gi = b_ih + [gi_base] + x @ W_ih[:, coff:coff+E].T   (small-K GEMM done as FMAs)
// gh comes from the WMMA GEMM (bias added here).
__global__ void gru_gates(
    const float* __restrict__ gh, const float* __restrict__ x,
    const float* __restrict__ W_ih, int ldw, int coff, int E,
    const float* __restrict__ b_ih, const float* __restrict__ b_hh,
    const float* __restrict__ gi_base,
    float* __restrict__ h, __bf16* __restrict__ h_bf, int B_, int H)
{
    int idx = blockIdx.x * blockDim.x + threadIdx.x;
    if (idx >= B_ * H) return;
    int b = idx / H, hh = idx % H;
    int jr = hh, jz = hh + H, jn = hh + 2 * H;

    float gir = b_ih[jr], giz = b_ih[jz], gin = b_ih[jn];
    if (gi_base) {
        const float* g = gi_base + (size_t)b * 3 * H;
        gir += g[jr]; giz += g[jz]; gin += g[jn];
    }
    const float* xb = x + (size_t)b * E;
    for (int k = 0; k < E; ++k) {
        float xv = xb[k];
        gir += xv * W_ih[(size_t)jr * ldw + coff + k];
        giz += xv * W_ih[(size_t)jz * ldw + coff + k];
        gin += xv * W_ih[(size_t)jn * ldw + coff + k];
    }
    const float* gb = gh + (size_t)b * 3 * H;
    float ghr = gb[jr] + b_hh[jr];
    float ghz = gb[jz] + b_hh[jz];
    float ghn = gb[jn] + b_hh[jn];

    float r = 1.f / (1.f + __expf(-(gir + ghr)));
    float z = 1.f / (1.f + __expf(-(giz + ghz)));
    float n = tanhf(gin + r * ghn);
    float hv = h[idx];
    float hn = (1.f - z) * n + z * hv;
    h[idx]    = hn;
    h_bf[idx] = (__bf16)hn;
}

// pred[b, t] = h[b,:] . W_dense + b_dense ; one wave (32 lanes) per batch row.
__global__ void dense_pred(const float* __restrict__ h, const float* __restrict__ Wd,
                           const float* __restrict__ bd, float* __restrict__ out,
                           int B_, int H, int T, int t)
{
    int w    = (blockIdx.x * blockDim.x + threadIdx.x) >> 5;
    int lane = threadIdx.x & 31;
    if (w >= B_) return;
    const float* hb = h + (size_t)w * H;
    float s = 0.f;
    for (int k = lane; k < H; k += 32) s += hb[k] * Wd[k];
#pragma unroll
    for (int off = 16; off > 0; off >>= 1) s += __shfl_down(s, off, 32);
    if (lane == 0) out[(size_t)w * T + t] = s + bd[0];
}

extern "C" void kernel_launch(void* const* d_in, const int* in_sizes, int n_in,
                              void* d_out, int out_size, void* d_ws, size_t ws_size,
                              hipStream_t stream) {
    (void)in_sizes; (void)n_in; (void)out_size; (void)ws_size;
    const float* lag     = (const float*)d_in[0];
    const float* curr    = (const float*)d_in[1];
    const float* W_ih_e  = (const float*)d_in[2];
    const float* W_hh_e  = (const float*)d_in[3];
    const float* b_ih_e  = (const float*)d_in[4];
    const float* b_hh_e  = (const float*)d_in[5];
    const float* W_ih_d  = (const float*)d_in[6];
    const float* W_hh_d  = (const float*)d_in[7];
    const float* b_ih_d  = (const float*)d_in[8];
    const float* b_hh_d  = (const float*)d_in[9];
    const float* W_dense = (const float*)d_in[10];
    const float* b_dense = (const float*)d_in[11];
    float* out = (float*)d_out;

    constexpr int T_enc = 168, T_dec = 24, B = 1024, E = 8, D = 16;
    constexpr int H = 512, H3 = 1536, LDWD = 528;

    char* wsp = (char*)d_ws;
    size_t off = 0;
    auto alloc = [&](size_t bytes) -> void* {
        off = (off + 255) & ~(size_t)255;
        void* p = wsp + off; off += bytes; return p;
    };
    __bf16* Whe_bf = (__bf16*)alloc((size_t)H3 * H * 2);   // W_hh_enc bf16
    __bf16* Whd_bf = (__bf16*)alloc((size_t)H3 * H * 2);   // W_hh_dec bf16
    __bf16* Wid_bf = (__bf16*)alloc((size_t)H3 * H * 2);   // W_ih_dec[:, :512] bf16
    float*  h_e    = (float*) alloc((size_t)B * H * 4);
    __bf16* h_e_bf = (__bf16*)alloc((size_t)B * H * 2);
    float*  h_d    = (float*) alloc((size_t)B * H * 4);
    __bf16* h_d_bf = (__bf16*)alloc((size_t)B * H * 2);
    float*  gh     = (float*) alloc((size_t)B * H3 * 4);   // per-step recurrent GEMM out
    float*  gid    = (float*) alloc((size_t)B * H3 * 4);   // h_t @ W_ih_dec[:, :512].T

    hipMemsetAsync(h_e,    0, (size_t)B * H * 4, stream);
    hipMemsetAsync(h_e_bf, 0, (size_t)B * H * 2, stream);
    hipMemsetAsync(h_d,    0, (size_t)B * H * 4, stream);
    hipMemsetAsync(h_d_bf, 0, (size_t)B * H * 2, stream);

    {
        int n = H3 * H, blk = (n + 255) / 256;
        cvt_bf16<<<blk, 256, 0, stream>>>(W_hh_e, Whe_bf, H3, H, H,    0);
        cvt_bf16<<<blk, 256, 0, stream>>>(W_hh_d, Whd_bf, H3, H, H,    0);
        cvt_bf16<<<blk, 256, 0, stream>>>(W_ih_d, Wid_bf, H3, H, LDWD, 0);
    }

    dim3 ggrid(H3 / 128, B / 128);          // 12 x 8 = 96 blocks
    int  eblk = (B * H + 255) / 256;        // 2048 blocks

    // Encoder: 168 sequential GRU steps
    for (int t = 0; t < T_enc; ++t) {
        gemm_nt_bf16<<<ggrid, 256, 0, stream>>>(h_e_bf, Whe_bf, gh, B, H3, H);
        gru_gates<<<eblk, 256, 0, stream>>>(gh, lag + (size_t)t * B * E,
                                            W_ih_e, E, 0, E, b_ih_e, b_hh_e,
                                            nullptr, h_e, h_e_bf, B, H);
    }

    // Decoder input part from h_t is constant across all 24 steps: one GEMM.
    gemm_nt_bf16<<<ggrid, 256, 0, stream>>>(h_e_bf, Wid_bf, gid, B, H3, H);

    // Decoder: 24 sequential GRU steps + dense head
    for (int t = 0; t < T_dec; ++t) {
        gemm_nt_bf16<<<ggrid, 256, 0, stream>>>(h_d_bf, Whd_bf, gh, B, H3, H);
        gru_gates<<<eblk, 256, 0, stream>>>(gh, curr + (size_t)t * B * D,
                                            W_ih_d, LDWD, H, D, b_ih_d, b_hh_d,
                                            gid, h_d, h_d_bf, B, H);
        dense_pred<<<(B * 32 + 255) / 256, 256, 0, stream>>>(h_d, W_dense, b_dense,
                                                             out, B, H, T_dec, t);
    }
}
